// GraphAttnEncoder_83708912599709
// MI455X (gfx1250) — matrix operand order repaired and checked by reading, not compiled, not run on hardware
//
#include <hip/hip_runtime.h>
#include <hip/hip_bf16.h>
#include <math.h>

// ---------------------------------------------------------------------------
// GraphAttnEncoder forward for MI455X (gfx1250, wave32, WMMA bf16).
// Dense GEMMs: v_wmma_f32_16x16x32_bf16, A-tile staged via async copy to LDS.
// Segment ops (softmax over dst, scatter-add) use f32 atomics.
// Workspace requirement: ~1.25 GB.
// ---------------------------------------------------------------------------

#define NNODE 50000
#define NEDGE 800000
#define DIM_D 128
#define DIM_E 64
#define NHEAD 4
#define HDIM  32
#define FFN_N 512
#define FFN_E 256
#define NLAYER 5

typedef __attribute__((ext_vector_type(16))) __bf16 v16bf;
typedef __attribute__((ext_vector_type(8)))  float  v8f;
typedef __attribute__((ext_vector_type(4)))  int    v4i;

union Frag16 { unsigned int u[8]; v16bf v; };

// ---- gfx1250 async global->LDS copy (guarded; sync fallback keeps compile OK)
#ifdef __has_builtin
#  if __has_builtin(__builtin_amdgcn_global_load_async_to_lds_b128)
#    define HAVE_ASYNC_LDS 1
#  endif
#  if __has_builtin(__builtin_amdgcn_s_wait_asynccnt)
#    define HAVE_WAIT_ASYNC 1
#  endif
#endif
#ifndef HAVE_ASYNC_LDS
#define HAVE_ASYNC_LDS 0
#endif
#ifndef HAVE_WAIT_ASYNC
#define HAVE_WAIT_ASYNC 0
#endif

#if HAVE_ASYNC_LDS
typedef __attribute__((address_space(1))) v4i* gv4i_p;   // global v4i*
typedef __attribute__((address_space(3))) v4i* lv4i_p;   // LDS v4i*

__device__ __forceinline__ void async_cp16(const void* g, void* l) {
  // global: flat address == AS(1) address (numeric round-trip is exact).
  // LDS: low 32 bits of a flat LDS pointer are the LDS byte offset.
  __builtin_amdgcn_global_load_async_to_lds_b128(
      (gv4i_p)(unsigned long long)g,
      (lv4i_p)(unsigned)(unsigned long long)l,
      0, 0);
}
__device__ __forceinline__ void wait_async0() {
#if HAVE_WAIT_ASYNC
  __builtin_amdgcn_s_wait_asynccnt(0);
#else
  asm volatile("s_wait_asynccnt 0" ::: "memory");
#endif
}
#endif

// pack two f32 -> dword of two bf16 (RNE via hardware cvt)
__device__ __forceinline__ unsigned pk2(float lo, float hi) {
  union { __bf16 h[2]; unsigned u; } t;
  t.h[0] = (__bf16)lo; t.h[1] = (__bf16)hi;
  return t.u;
}
__device__ __forceinline__ unsigned short bf1(float f) {
  union { __bf16 h; unsigned short u; } t;
  t.h = (__bf16)f;
  return t.u;
}

__device__ __forceinline__ void atomicMaxF(float* addr, float v) {
  int iv = __float_as_int(v);
  if (iv >= 0) atomicMax((int*)addr, iv);
  else         atomicMin((unsigned int*)addr, (unsigned int)iv);
}

// ---------------------------------------------------------------------------
// WMMA GEMM: Y[M,Nn] = act(X[M,K] @ W[K,Nn] + bias)
// block = 256 threads (8 waves), tile 64x64, K-step 32.
// A tile staged in native precision (async b128 copies), converted to bf16 at
// fragment build. B tile converted to bf16 at staging (reused by 4 waves).
// INBF: X is bf16 (raw u16).  OUTBF: write bf16.  RELU: apply relu.
// ---------------------------------------------------------------------------
template<bool INBF, bool OUTBF, bool RELU>
__global__ __launch_bounds__(256) void gemm_wmma_kernel(
    const void* __restrict__ Xv, const float* __restrict__ W,
    const float* __restrict__ bias, void* __restrict__ Yv,
    int M, int K, int Nn) {
  // A tile: 64 rows x 32 K. bf16 rows padded to 80B, f32 rows to 144B
  // (both 16B-aligned row stride; bank-spread for fragment reads).
  __shared__ alignas(16) unsigned char Abuf[INBF ? 64 * 80 : 64 * 144];
  __shared__ unsigned short Bs[32][66];   // 32 K x 64 N bf16, +2 pad
  const int tid  = threadIdx.x;
  const int bm0  = blockIdx.x * 64;
  const int bn0  = blockIdx.y * 64;
  const int lane = tid & 31;
  const int wave = tid >> 5;
  const int wm   = wave & 3;      // 4 wave rows (16 each)
  const int wn   = wave >> 2;     // 2 wave cols (32 each)
  v8f c0 = {}; v8f c1 = {};
  const int bk  = tid >> 3;            // B staging: K row
  const int bnc = (tid & 7) * 8;       // B staging: 8 consecutive N

  for (int k0 = 0; k0 < K; k0 += 32) {
    // ---- stage A (row-clamped: no branches; garbage rows masked at store)
    if (INBF) {
      const unsigned short* X = (const unsigned short*)Xv;
      const int c = tid;               // 256 chunks of 16B, 4 per 64B row
      const int r = c >> 2, cp = c & 3;
      int grow = bm0 + r; if (grow >= M) grow = M - 1;
      const void* gsrc = X + (size_t)grow * K + k0 + cp * 8;
      void* ldst = Abuf + r * 80 + cp * 16;
#if HAVE_ASYNC_LDS
      async_cp16(gsrc, ldst);
#else
      *(uint4*)ldst = *(const uint4*)gsrc;
#endif
    } else {
      const float* X = (const float*)Xv;
      #pragma unroll
      for (int q = 0; q < 2; ++q) {
        const int c = tid + q * 256;   // 512 chunks of 16B, 8 per 128B row
        const int r = c >> 3, cp = c & 7;
        int grow = bm0 + r; if (grow >= M) grow = M - 1;
        const void* gsrc = X + (size_t)grow * K + k0 + cp * 4;
        void* ldst = Abuf + r * 144 + cp * 16;
#if HAVE_ASYNC_LDS
        async_cp16(gsrc, ldst);
#else
        *(uint4*)ldst = *(const uint4*)gsrc;
#endif
      }
    }
    // ---- stage B (f32 -> bf16 once; vectorized)
    {
      const float* Wp = W + (size_t)(k0 + bk) * Nn + bn0 + bnc;
      float4 w0 = *(const float4*)(Wp);
      float4 w1 = *(const float4*)(Wp + 4);
      unsigned* Bd = (unsigned*)&Bs[bk][bnc];
      Bd[0] = pk2(w0.x, w0.y);
      Bd[1] = pk2(w0.z, w0.w);
      Bd[2] = pk2(w1.x, w1.y);
      Bd[3] = pk2(w1.z, w1.w);
    }
#if HAVE_ASYNC_LDS
    wait_async0();
#endif
    __syncthreads();

    // ---- fragments
    // A: lane -> M (lane&15); K split: lanes0-15 K 0-7/16-23,
    //    lanes16-31 K 8-15/24-31, pairs per dword.
    Frag16 a, b0, b1;
    const int r  = wm * 16 + (lane & 15);
    const int kb = (lane >> 4) << 3;
    if (INBF) {
      const unsigned short* Ar = (const unsigned short*)(Abuf + r * 80);
      #pragma unroll
      for (int p = 0; p < 4; ++p) {
        a.u[p]     = *(const unsigned int*)&Ar[kb + 2 * p];
        a.u[4 + p] = *(const unsigned int*)&Ar[16 + kb + 2 * p];
      }
    } else {
      const float* Ar = (const float*)(Abuf + r * 144);
      #pragma unroll
      for (int p = 0; p < 4; ++p) {
        a.u[p]     = pk2(Ar[kb + 2 * p],      Ar[kb + 2 * p + 1]);
        a.u[4 + p] = pk2(Ar[16 + kb + 2 * p], Ar[16 + kb + 2 * p + 1]);
      }
    }
    // B: lane -> N (lane&15), K split same as A.
    const int bn = wn * 32 + (lane & 15);
    #pragma unroll
    for (int p = 0; p < 4; ++p) {
      b0.u[p]     = (unsigned)Bs[kb + 2 * p][bn] | ((unsigned)Bs[kb + 2 * p + 1][bn] << 16);
      b0.u[4 + p] = (unsigned)Bs[16 + kb + 2 * p][bn] | ((unsigned)Bs[16 + kb + 2 * p + 1][bn] << 16);
      b1.u[p]     = (unsigned)Bs[kb + 2 * p][bn + 16] | ((unsigned)Bs[kb + 2 * p + 1][bn + 16] << 16);
      b1.u[4 + p] = (unsigned)Bs[16 + kb + 2 * p][bn + 16] | ((unsigned)Bs[16 + kb + 2 * p + 1][bn + 16] << 16);
    }
    c0 = __builtin_amdgcn_wmma_f32_16x16x32_bf16(false, a.v, false, b0.v,
                                                 (short)0, c0, false, false);
    c1 = __builtin_amdgcn_wmma_f32_16x16x32_bf16(false, a.v, false, b1.v,
                                                 (short)0, c1, false, false);
    __syncthreads();
  }

  // C/D layout: VGPR i -> M=i (lanes 0-15) or M=8+i (lanes 16-31); lane&15 -> N.
  const int ncol  = bn0 + wn * 32 + (lane & 15);
  const int mbase = bm0 + wm * 16 + ((lane >> 4) << 3);
  const float bv0 = bias ? bias[ncol] : 0.f;
  const float bv1 = bias ? bias[ncol + 16] : 0.f;
  #pragma unroll
  for (int i = 0; i < 8; ++i) {
    const int row = mbase + i;
    if (row < M) {
      float v0 = c0[i] + bv0;
      float v1 = c1[i] + bv1;
      if (RELU) { v0 = fmaxf(v0, 0.f); v1 = fmaxf(v1, 0.f); }
      if (OUTBF) {
        unsigned short* Y = (unsigned short*)Yv;
        Y[(size_t)row * Nn + ncol]      = bf1(v0);
        Y[(size_t)row * Nn + ncol + 16] = bf1(v1);
      } else {
        float* Y = (float*)Yv;
        Y[(size_t)row * Nn + ncol]      = v0;
        Y[(size_t)row * Nn + ncol + 16] = v1;
      }
    }
  }
}

// ---------------------------------------------------------------------------
// Elementwise / segment kernels
// ---------------------------------------------------------------------------
__global__ void fill_kernel(float* p, int n, float v) {
  int i = blockIdx.x * blockDim.x + threadIdx.x;
  if (i < n) p[i] = v;
}

__global__ void cnt_kernel(const int* __restrict__ dst, float* cnt, int n) {
  int i = blockIdx.x * blockDim.x + threadIdx.x;
  if (i < n) atomicAdd(&cnt[dst[i]], 1.0f);
}

__global__ void loopsum_kernel(const int* __restrict__ dst,
                               const float* __restrict__ ev, float* lsum, int n) {
  int i = blockIdx.x * blockDim.x + threadIdx.x;
  if (i < n) {
    int e = i >> 6, d = i & 63;
    atomicAdd(&lsum[(size_t)dst[e] * DIM_E + d], ev[i]);
  }
}

__global__ void loopdiv_kernel(float* lsum, const float* __restrict__ cnt, int n) {
  int i = blockIdx.x * blockDim.x + threadIdx.x;
  if (i < n) lsum[i] /= fmaxf(cnt[i >> 6], 1.0f);
}

// a_src / a_dst : per (node, head) dot of 32
__global__ void asd_kernel(const float* __restrict__ xh,
                           const float* __restrict__ attS,
                           const float* __restrict__ attD,
                           float* as_, float* ad_, int n) {
  int i = blockIdx.x * blockDim.x + threadIdx.x;
  if (i >= n) return;
  int nd = i >> 2, h = i & 3;
  const float* xr = xh + (size_t)nd * DIM_D + h * HDIM;
  const float* s = attS + h * HDIM;
  const float* d = attD + h * HDIM;
  float s1 = 0.f, s2 = 0.f;
  #pragma unroll
  for (int c = 0; c < HDIM; ++c) { s1 += xr[c] * s[c]; s2 += xr[c] * d[c]; }
  as_[i] = s1; ad_[i] = s2;
}

// Ve[d][h] = sum_c lin_edge_W[d, h*32+c] * att_edge[h,c]   (64x4)
__global__ void ve_kernel(const float* __restrict__ linEW,
                          const float* __restrict__ attE, float* Ve) {
  int t = threadIdx.x;                // 256 = 64*4
  int d = t >> 2, h = t & 3;
  float s = 0.f;
  #pragma unroll
  for (int c = 0; c < HDIM; ++c)
    s += linEW[(size_t)d * DIM_D + h * HDIM + c] * attE[h * HDIM + c];
  Ve[d * NHEAD + h] = s;
}

__global__ void alpha_kernel(const int* __restrict__ src, const int* __restrict__ dst,
                             const float* __restrict__ ev, const float* __restrict__ lea,
                             const float* __restrict__ Ve,
                             const float* __restrict__ as_, const float* __restrict__ ad_,
                             float* alpha, float* mx, int total) {
  int f = blockIdx.x * blockDim.x + threadIdx.x;
  if (f >= total) return;
  int sf, df; const float* row;
  if (f < NEDGE) { sf = src[f]; df = dst[f]; row = ev + (size_t)f * DIM_E; }
  else { sf = df = f - NEDGE; row = lea + (size_t)(f - NEDGE) * DIM_E; }
  float ae[4] = {0.f, 0.f, 0.f, 0.f};
  #pragma unroll 8
  for (int d = 0; d < DIM_E; ++d) {
    float r = row[d];
    const float* v = Ve + d * NHEAD;
    ae[0] += r * v[0]; ae[1] += r * v[1]; ae[2] += r * v[2]; ae[3] += r * v[3];
  }
  #pragma unroll
  for (int h = 0; h < NHEAD; ++h) {
    float a = as_[sf * NHEAD + h] + ad_[df * NHEAD + h] + ae[h];
    a = (a >= 0.f) ? a : 0.2f * a;                 // leaky_relu 0.2
    alpha[(size_t)f * NHEAD + h] = a;
    atomicMaxF(&mx[df * NHEAD + h], a);
  }
}

__global__ void exp_kernel(const int* __restrict__ dst, float* alpha,
                           const float* __restrict__ mx, float* den, int n) {
  int i = blockIdx.x * blockDim.x + threadIdx.x;
  if (i >= n) return;
  int f = i >> 2, h = i & 3;
  int df = (f < NEDGE) ? dst[f] : f - NEDGE;
  float e = __expf(alpha[i] - mx[df * NHEAD + h]);
  alpha[i] = e;
  atomicAdd(&den[df * NHEAD + h], e);
}

__global__ void scatter_kernel(const int* __restrict__ src, const int* __restrict__ dst,
                               const float* __restrict__ ex, const float* __restrict__ den,
                               const float* __restrict__ xh, float* accum, int n) {
  int i = blockIdx.x * blockDim.x + threadIdx.x;
  if (i >= n) return;
  int f = i >> 7, d = i & 127;
  int h = d >> 5;
  int sf, df;
  if (f < NEDGE) { sf = src[f]; df = dst[f]; } else { sf = df = f - NEDGE; }
  float w = ex[(size_t)f * NHEAD + h] / den[df * NHEAD + h];
  atomicAdd(&accum[(size_t)df * DIM_D + d], xh[(size_t)sf * DIM_D + d] * w);
}

__global__ void nubias_kernel(float* accum, const float* __restrict__ gatb, int n) {
  int i = blockIdx.x * blockDim.x + threadIdx.x;
  if (i < n) accum[i] += gatb[i & (DIM_D - 1)];
}

__global__ void edgeupd_kernel(const int* __restrict__ src, const int* __restrict__ dst,
                               const float* __restrict__ eun, const float* __restrict__ evIn,
                               float* eu, float* ev2, int n) {
  int i = blockIdx.x * blockDim.x + threadIdx.x;
  if (i >= n) return;
  int e = i >> 6, d = i & 63;
  float v = eun[(size_t)src[e] * DIM_E + d] + eun[(size_t)dst[e] * DIM_E + d];
  eu[i] = v;
  ev2[i] = evIn[i] + v;
}

// out = LayerNorm(A + B) * g + be ; one wave per row
__global__ __launch_bounds__(256) void ln_kernel(
    const float* __restrict__ A, const float* __restrict__ B,
    const float* __restrict__ g, const float* __restrict__ be,
    float* out, int M, int Dim) {
  int lane = threadIdx.x & 31;
  int wid  = threadIdx.x >> 5;
  int row  = blockIdx.x * 8 + wid;
  if (row >= M) return;
  size_t base = (size_t)row * Dim;
  int per = Dim >> 5;        // 4 (D=128) or 2 (De=64)
  float x[4];
  float s = 0.f;
  for (int j = 0; j < per; ++j) {
    int c = j * 32 + lane;
    x[j] = A[base + c] + B[base + c];
    s += x[j];
  }
  #pragma unroll
  for (int m = 16; m > 0; m >>= 1) s += __shfl_xor(s, m, 32);
  float mean = s / (float)Dim;
  float v = 0.f;
  for (int j = 0; j < per; ++j) { float d = x[j] - mean; v += d * d; }
  #pragma unroll
  for (int m = 16; m > 0; m >>= 1) v += __shfl_xor(v, m, 32);
  float inv = rsqrtf(v / (float)Dim + 1e-5f);
  for (int j = 0; j < per; ++j) {
    int c = j * 32 + lane;
    out[base + c] = (x[j] - mean) * inv * g[c] + be[c];
  }
}

// ---------------------------------------------------------------------------
// Host orchestration
// ---------------------------------------------------------------------------
static inline int nblk(long long n) { return (int)((n + 255) / 256); }

extern "C" void kernel_launch(void* const* d_in, const int* in_sizes, int n_in,
                              void* d_out, int out_size, void* d_ws, size_t ws_size,
                              hipStream_t stream) {
  (void)in_sizes; (void)n_in; (void)out_size; (void)ws_size;
  const float* node_vec = (const float*)d_in[0];
  const int*   edge_idx = (const int*)d_in[1];
  const float* edge_vec = (const float*)d_in[2];
  const float* lin_W    = (const float*)d_in[3];
  const float* att_src  = (const float*)d_in[4];
  const float* att_dst  = (const float*)d_in[5];
  const float* att_edge = (const float*)d_in[6];
  const float* lin_eW   = (const float*)d_in[7];
  const float* gat_b    = (const float*)d_in[8];
  const float* n2e_W    = (const float*)d_in[9];
  const float* n2e_b    = (const float*)d_in[10];
  const float* nW1      = (const float*)d_in[11];
  const float* nb1      = (const float*)d_in[12];
  const float* nW2      = (const float*)d_in[13];
  const float* nb2      = (const float*)d_in[14];
  const float* nln1_g   = (const float*)d_in[15];
  const float* nln1_b   = (const float*)d_in[16];
  const float* nln2_g   = (const float*)d_in[17];
  const float* nln2_b   = (const float*)d_in[18];
  const float* eW1      = (const float*)d_in[19];
  const float* eb1      = (const float*)d_in[20];
  const float* eW2      = (const float*)d_in[21];
  const float* eb2      = (const float*)d_in[22];
  const float* eln1_g   = (const float*)d_in[23];
  const float* eln1_b   = (const float*)d_in[24];
  const float* eln2_g   = (const float*)d_in[25];
  const float* eln2_b   = (const float*)d_in[26];

  const int* src = edge_idx;
  const int* dst = edge_idx + NEDGE;

  // ---- workspace carve (floats) ----
  float* w = (float*)d_ws;
  size_t off = 0;
  auto take = [&](size_t n) { float* r = w + off; off += n; return r; };
  float* nbuf0 = take((size_t)NNODE * DIM_D);
  float* nbuf1 = take((size_t)NNODE * DIM_D);
  float* nbuf[2] = { nbuf0, nbuf1 };
  float* ebuf[3];
  ebuf[0] = take((size_t)NEDGE * DIM_E);
  ebuf[1] = take((size_t)NEDGE * DIM_E);
  ebuf[2] = take((size_t)NEDGE * DIM_E);
  float* xh    = take((size_t)NNODE * DIM_D);   // also reused as h2
  float* asrc  = take((size_t)NNODE * NHEAD);
  float* adst  = take((size_t)NNODE * NHEAD);
  float* Ve    = take((size_t)DIM_E * NHEAD);
  float* alpha = take((size_t)(NEDGE + NNODE) * NHEAD);
  float* mx    = take((size_t)NNODE * NHEAD);
  float* den   = take((size_t)NNODE * NHEAD);
  float* cnt   = take((size_t)NNODE);
  float* lea   = take((size_t)NNODE * DIM_E);
  float* accum = take((size_t)NNODE * DIM_D);   // GAT out -> nu
  float* eun   = take((size_t)NNODE * DIM_E);
  float* x1    = take((size_t)NNODE * DIM_D);
  unsigned short* h1bf = (unsigned short*)take((size_t)NNODE * FFN_N / 2);
  unsigned short* hebf = (unsigned short*)take((size_t)NEDGE * FFN_E / 2);

  (void)hipMemcpyAsync(nbuf[0], node_vec, (size_t)NNODE * DIM_D * sizeof(float),
                       hipMemcpyDeviceToDevice, stream);
  (void)hipMemcpyAsync(ebuf[0], edge_vec, (size_t)NEDGE * DIM_E * sizeof(float),
                       hipMemcpyDeviceToDevice, stream);

  int ncur = 0, ecur = 0;
  for (int l = 0; l < NLAYER; ++l) {
    const float* p_linW  = lin_W   + (size_t)l * DIM_D * DIM_D;
    const float* p_attS  = att_src + (size_t)l * NHEAD * HDIM;
    const float* p_attD  = att_dst + (size_t)l * NHEAD * HDIM;
    const float* p_attE  = att_edge+ (size_t)l * NHEAD * HDIM;
    const float* p_linEW = lin_eW  + (size_t)l * DIM_E * DIM_D;
    const float* p_gatb  = gat_b   + (size_t)l * DIM_D;
    const float* p_n2eW  = n2e_W   + (size_t)l * DIM_D * DIM_E;
    const float* p_n2eb  = n2e_b   + (size_t)l * DIM_E;
    const float* p_nW1   = nW1     + (size_t)l * DIM_D * FFN_N;
    const float* p_nb1   = nb1     + (size_t)l * FFN_N;
    const float* p_nW2   = nW2     + (size_t)l * FFN_N * DIM_D;
    const float* p_nb2   = nb2     + (size_t)l * DIM_D;
    const float* p_nl1g  = nln1_g  + (size_t)l * DIM_D;
    const float* p_nl1b  = nln1_b  + (size_t)l * DIM_D;
    const float* p_nl2g  = nln2_g  + (size_t)l * DIM_D;
    const float* p_nl2b  = nln2_b  + (size_t)l * DIM_D;
    const float* p_eW1   = eW1     + (size_t)l * DIM_E * FFN_E;
    const float* p_eb1   = eb1     + (size_t)l * FFN_E;
    const float* p_eW2   = eW2     + (size_t)l * FFN_E * DIM_E;
    const float* p_eb2   = eb2     + (size_t)l * DIM_E;
    const float* p_el1g  = eln1_g  + (size_t)l * DIM_E;
    const float* p_el1b  = eln1_b  + (size_t)l * DIM_E;
    const float* p_el2g  = eln2_g  + (size_t)l * DIM_E;
    const float* p_el2b  = eln2_b  + (size_t)l * DIM_E;

    float* nvIn  = nbuf[ncur];
    float* nvOut = nbuf[1 - ncur];
    float* evIn  = ebuf[ecur];             // later reused as xe
    float* eu    = ebuf[(ecur + 1) % 3];   // later reused as he2
    float* ev2   = ebuf[(ecur + 2) % 3];   // later reused as evOut

    // ---- GAT ----
    fill_kernel<<<nblk(NNODE), 256, 0, stream>>>(cnt, NNODE, 0.f);
    fill_kernel<<<nblk((long long)NNODE * DIM_E), 256, 0, stream>>>(lea, NNODE * DIM_E, 0.f);
    fill_kernel<<<nblk(NNODE * NHEAD), 256, 0, stream>>>(mx, NNODE * NHEAD, -INFINITY);
    fill_kernel<<<nblk(NNODE * NHEAD), 256, 0, stream>>>(den, NNODE * NHEAD, 0.f);
    fill_kernel<<<nblk((long long)NNODE * DIM_D), 256, 0, stream>>>(accum, NNODE * DIM_D, 0.f);

    cnt_kernel<<<nblk(NEDGE), 256, 0, stream>>>(dst, cnt, NEDGE);
    loopsum_kernel<<<nblk((long long)NEDGE * DIM_E), 256, 0, stream>>>(dst, evIn, lea, NEDGE * DIM_E);
    loopdiv_kernel<<<nblk((long long)NNODE * DIM_E), 256, 0, stream>>>(lea, cnt, NNODE * DIM_E);

    gemm_wmma_kernel<false, false, false>
        <<<dim3((NNODE + 63) / 64, DIM_D / 64), 256, 0, stream>>>(
            nvIn, p_linW, nullptr, xh, NNODE, DIM_D, DIM_D);

    asd_kernel<<<nblk(NNODE * NHEAD), 256, 0, stream>>>(xh, p_attS, p_attD, asrc, adst, NNODE * NHEAD);
    ve_kernel<<<1, 256, 0, stream>>>(p_linEW, p_attE, Ve);
    alpha_kernel<<<nblk(NEDGE + NNODE), 256, 0, stream>>>(
        src, dst, evIn, lea, Ve, asrc, adst, alpha, mx, NEDGE + NNODE);
    exp_kernel<<<nblk((long long)(NEDGE + NNODE) * NHEAD), 256, 0, stream>>>(
        dst, alpha, mx, den, (NEDGE + NNODE) * NHEAD);
    scatter_kernel<<<nblk((long long)(NEDGE + NNODE) * DIM_D), 256, 0, stream>>>(
        src, dst, alpha, den, xh, accum, (NEDGE + NNODE) * DIM_D);
    nubias_kernel<<<nblk((long long)NNODE * DIM_D), 256, 0, stream>>>(accum, p_gatb, NNODE * DIM_D);

    // ---- node->edge update ----
    gemm_wmma_kernel<false, false, false>
        <<<dim3((NNODE + 63) / 64, DIM_E / 64), 256, 0, stream>>>(
            accum, p_n2eW, p_n2eb, eun, NNODE, DIM_D, DIM_E);
    edgeupd_kernel<<<nblk((long long)NEDGE * DIM_E), 256, 0, stream>>>(
        src, dst, eun, evIn, eu, ev2, NEDGE * DIM_E);

    // ---- node FFN block ----
    ln_kernel<<<(NNODE + 7) / 8, 256, 0, stream>>>(nvIn, accum, p_nl1g, p_nl1b, x1, NNODE, DIM_D);
    gemm_wmma_kernel<false, true, true>
        <<<dim3((NNODE + 63) / 64, FFN_N / 64), 256, 0, stream>>>(
            x1, p_nW1, p_nb1, h1bf, NNODE, DIM_D, FFN_N);
    gemm_wmma_kernel<true, false, false>
        <<<dim3((NNODE + 63) / 64, DIM_D / 64), 256, 0, stream>>>(
            h1bf, p_nW2, p_nb2, xh /* h2 */, NNODE, FFN_N, DIM_D);
    ln_kernel<<<(NNODE + 7) / 8, 256, 0, stream>>>(x1, xh, p_nl2g, p_nl2b, nvOut, NNODE, DIM_D);

    // ---- edge FFN block (dominant compute: E x 64 x 256 x2) ----
    ln_kernel<<<(NEDGE + 7) / 8, 256, 0, stream>>>(ev2, eu, p_el1g, p_el1b, evIn /* xe */, NEDGE, DIM_E);
    gemm_wmma_kernel<false, true, true>
        <<<dim3((NEDGE + 63) / 64, FFN_E / 64), 256, 0, stream>>>(
            evIn, p_eW1, p_eb1, hebf, NEDGE, DIM_E, FFN_E);
    gemm_wmma_kernel<true, false, false>
        <<<dim3((NEDGE + 63) / 64, DIM_E / 64), 256, 0, stream>>>(
            hebf, p_eW2, p_eb2, eu /* he2 */, NEDGE, FFN_E, DIM_E);
    ln_kernel<<<(NEDGE + 7) / 8, 256, 0, stream>>>(evIn, eu, p_el2g, p_el2b, ev2 /* evOut */, NEDGE, DIM_E);

    ncur ^= 1;
    ecur = (ecur + 2) % 3;
  }

  (void)hipMemcpyAsync(d_out, nbuf[ncur], (size_t)NNODE * DIM_D * sizeof(float),
                       hipMemcpyDeviceToDevice, stream);
  (void)hipMemcpyAsync((float*)d_out + (size_t)NNODE * DIM_D, ebuf[ecur],
                       (size_t)NEDGE * DIM_E * sizeof(float),
                       hipMemcpyDeviceToDevice, stream);
}
